// MultiheadSelfAttention_12876311953588
// MI455X (gfx1250) — matrix operand order repaired and checked by reading, not compiled, not run on hardware
//
#include <hip/hip_runtime.h>
#include <math.h>

// ---- problem constants (fixed by reference) ----
#define S_LEN   2048
#define D_MODEL 1024
#define NHEADS  16
#define DHEAD   64
#define BATCH   4
#define M_ROWS  (BATCH * S_LEN)   // 8192

typedef __attribute__((ext_vector_type(16))) __bf16 bf16x16;
typedef __attribute__((ext_vector_type(8)))  float  f32x8;

union FragU { uint4 q[2]; bf16x16 v; };

__device__ __forceinline__ unsigned short f2bf(float f) {
    unsigned u = __builtin_bit_cast(unsigned, f);
    unsigned r = (u + 0x7FFFu + ((u >> 16) & 1u)) >> 16;   // round-to-nearest-even
    return (unsigned short)r;
}
__device__ __forceinline__ float bf2f(unsigned short h) {
    return __builtin_bit_cast(float, ((unsigned)h) << 16);
}
__device__ __forceinline__ unsigned pack2(float a, float b) {
    return (unsigned)f2bf(a) | ((unsigned)f2bf(b) << 16);
}
__device__ __forceinline__ f32x8 wmma_bf16(bf16x16 a, bf16x16 b, f32x8 c) {
    return __builtin_amdgcn_wmma_f32_16x16x32_bf16(false, a, false, b, (short)0, c, false, false);
}

// A-fragment: 16(M) x 32(K) bf16 from row-major [m][k] (stride in elements).
// Contiguous k order (k = 16*half + s): lanes 0-15 hold M=lane, K 0-15;
//             lanes 16-31 hold K 16-31.
__device__ __forceinline__ FragU loadA(const unsigned short* base, int stride, int lane) {
    int m = lane & 15, hf = lane >> 4;
    FragU f;
    f.q[0] = *(const uint4*)(base + (size_t)m * stride + hf * 16);
    f.q[1] = *(const uint4*)(base + (size_t)m * stride + hf * 16 + 8);
    return f;
}
// B-fragment: 32(K) x 16(N) bf16 where B[k][n] = Mrow[n][k] (Mrow row-major, stride elems).
// Contiguous k order, same as loadA: lanes 0-15: N=lane, K 0-15; lanes 16-31: N=lane-16, K 16-31.
__device__ __forceinline__ FragU loadB(const unsigned short* base, int stride, int lane) {
    int n = lane & 15, hf = lane >> 4;
    FragU f;
    f.q[0] = *(const uint4*)(base + (size_t)n * stride + hf * 16);
    f.q[1] = *(const uint4*)(base + (size_t)n * stride + hf * 16 + 8);
    return f;
}

// ---------------- fp32 -> bf16 convert ----------------
__global__ void cvt_f32_bf16(const float* __restrict__ src,
                             unsigned short* __restrict__ dst, int n4) {
    int i = blockIdx.x * blockDim.x + threadIdx.x;
    if (i < n4) {
        float4 f = ((const float4*)src)[i];
        uint2 o;
        o.x = pack2(f.x, f.y);
        o.y = pack2(f.z, f.w);
        ((uint2*)dst)[i] = o;
    }
}

// ---------------- tiled WMMA GEMM:  Y[m,n] = sum_k A[m,k] * W[n,k] ----------------
// MODE 0: write bf16 to [B,H,S,Dh];  MODE 1: write bf16 to [B,H,Dh,S] (V-transpose);
// MODE 2: write fp32 row-major [M, 1024]
#define LDSS 40   // padded bf16 stride: 80B = 20 banks -> 16 rows conflict-free, 16B aligned
template <int MODE>
__global__ __launch_bounds__(256) void gemm_bf16(
    const unsigned short* __restrict__ A,   // [M_ROWS, 1024] bf16
    const unsigned short* __restrict__ W,   // [1024, 1024]   bf16 (row n = output feature)
    void* __restrict__ out)
{
    const int K = D_MODEL;
    __shared__ unsigned short lA[128 * LDSS];
    __shared__ unsigned short lB[128 * LDSS];

    int tid  = threadIdx.x;
    int lane = tid & 31, wave = tid >> 5;
    int wr = wave >> 1, wc = wave & 1;         // 4x2 waves; each wave: 32 rows x 64 cols
    int m0 = blockIdx.y * 128, n0 = blockIdx.x * 128;

    f32x8 acc[2][4];
#pragma unroll
    for (int i = 0; i < 2; i++)
#pragma unroll
        for (int j = 0; j < 4; j++) acc[i][j] = (f32x8)(0.0f);

    int r = tid >> 2, seg = tid & 3;
    for (int kk = 0; kk < K; kk += 32) {
#pragma unroll
        for (int p = 0; p < 2; p++) {
            int row = p * 64 + r;
            *(uint4*)(&lA[row * LDSS + seg * 8]) =
                *(const uint4*)(A + (size_t)(m0 + row) * K + kk + seg * 8);
            *(uint4*)(&lB[row * LDSS + seg * 8]) =
                *(const uint4*)(W + (size_t)(n0 + row) * K + kk + seg * 8);
        }
        __syncthreads();

        FragU af[2], bf[4];
#pragma unroll
        for (int i = 0; i < 2; i++) af[i] = loadA(lA + (wr * 32 + i * 16) * LDSS, LDSS, lane);
#pragma unroll
        for (int j = 0; j < 4; j++) bf[j] = loadB(lB + (wc * 64 + j * 16) * LDSS, LDSS, lane);

#pragma unroll
        for (int i = 0; i < 2; i++)
#pragma unroll
            for (int j = 0; j < 4; j++)
                acc[i][j] = wmma_bf16(af[i].v, bf[j].v, acc[i][j]);
        __syncthreads();
    }

    // epilogue: C layout -> lane owns column n = (lane&15), rows M split 0-7 / 8-15 by half
    int hi = lane >> 4;
#pragma unroll
    for (int i = 0; i < 2; i++) {
#pragma unroll
        for (int j = 0; j < 4; j++) {
            int gn = n0 + wc * 64 + j * 16 + (lane & 15);
#pragma unroll
            for (int r8 = 0; r8 < 8; r8++) {
                int gm = m0 + wr * 32 + i * 16 + hi * 8 + r8;
                float v = acc[i][j][r8];
                if constexpr (MODE == 2) {
                    ((float*)out)[(size_t)gm * D_MODEL + gn] = v;
                } else {
                    int b = gm >> 11, s = gm & (S_LEN - 1);
                    int h = gn >> 6,  d = gn & (DHEAD - 1);
                    if constexpr (MODE == 0)
                        ((unsigned short*)out)[(((size_t)(b * NHEADS + h) * S_LEN + s) << 6) + d] = f2bf(v);
                    else
                        ((unsigned short*)out)[(((size_t)(b * NHEADS + h) * DHEAD + d) << 11) + s] = f2bf(v);
                }
            }
        }
    }
}

// ---------------- RoPE (in place, bf16 [B,H,S,Dh]); 'scale' folds 1/sqrt(Dh) for Q ----------------
__global__ void rope_kernel(unsigned short* __restrict__ t,
                            const int* __restrict__ pos, float scale) {
    int idx = blockIdx.x * blockDim.x + threadIdx.x;     // B*H*S*(Dh/2) = 4M
    if (idx >= BATCH * NHEADS * S_LEN * (DHEAD / 2)) return;
    int i  = idx & 31;
    int s  = (idx >> 5) & (S_LEN - 1);
    int bh = idx >> 16;
    size_t off = ((size_t)bh * S_LEN + s) * DHEAD + 2 * i;
    float p   = (float)pos[s];
    // inv_freq = exp(-i * 2/64 * ln(10000)) ; 2/64*ln(1e4) = 0.28782313662...
    float ang = p * __expf(-0.28782313662425572f * (float)i);
    float sn, cs;
    __sincosf(ang, &sn, &cs);
    float te = bf2f(t[off]), to = bf2f(t[off + 1]);
    float oe = (te * cs - to * sn) * scale;
    float oo = (te * sn + to * cs) * scale;
    *(unsigned*)(t + off) = pack2(oe, oo);
}

// ---------------- flash attention: one wave per (b,h, 32-row q tile) ----------------
// S^T = K x Q^T (softmax rows land in C-layout M direction), O^T += V^T x P
__global__ __launch_bounds__(32) void flash_attn(
    const unsigned short* __restrict__ qb,   // [B*H, S, 64]  (pre-scaled by 1/8)
    const unsigned short* __restrict__ kb,   // [B*H, S, 64]
    const unsigned short* __restrict__ vtb,  // [B*H, 64, S]
    unsigned short* __restrict__ attn)       // [B, S, 1024] bf16
{
    int lane = threadIdx.x;
    int qt = blockIdx.x;            // 32-row q tile, 0..63
    int bh = blockIdx.y;            // 0..63
    int hi = lane >> 4, l15 = lane & 15;

    const unsigned short* qp = qb  + ((size_t)bh * S_LEN + qt * 32) * DHEAD;
    const unsigned short* kp = kb  + (size_t)bh * S_LEN * DHEAD;
    const unsigned short* vp = vtb + (size_t)bh * DHEAD * S_LEN;

    FragU bq[2][2];                 // Q as B-operand: [q-subtile][d-step]
#pragma unroll
    for (int j = 0; j < 2; j++)
#pragma unroll
        for (int s = 0; s < 2; s++)
            bq[j][s] = loadB(qp + j * 16 * DHEAD + s * 32, DHEAD, lane);

    f32x8 ot[4][2];                 // O^T accum: [d-subtile][q-subtile]
#pragma unroll
    for (int a = 0; a < 4; a++)
#pragma unroll
        for (int j = 0; j < 2; j++) ot[a][j] = (f32x8)(0.0f);
    float mrow[2] = { -__builtin_inff(), -__builtin_inff() };
    float lrow[2] = { 0.0f, 0.0f };

    for (int kt = 0; kt <= qt; kt++) {
        FragU ak[2][2];
#pragma unroll
        for (int i = 0; i < 2; i++)
#pragma unroll
            for (int s = 0; s < 2; s++)
                ak[i][s] = loadA(kp + (size_t)(kt * 32 + i * 16) * DHEAD + s * 32, DHEAD, lane);

        f32x8 st[2][2];             // S^T tile: [key-subtile][q-subtile]
#pragma unroll
        for (int i = 0; i < 2; i++)
#pragma unroll
            for (int j = 0; j < 2; j++) {
                st[i][j] = wmma_bf16(ak[i][0].v, bq[j][0].v, (f32x8)(0.0f));
                st[i][j] = wmma_bf16(ak[i][1].v, bq[j][1].v, st[i][j]);
            }

        if (kt == qt) {             // causal mask on diagonal tile only
#pragma unroll
            for (int i = 0; i < 2; i++)
#pragma unroll
                for (int j = 0; j < 2; j++)
#pragma unroll
                    for (int r = 0; r < 8; r++) {
                        int key = kt * 32 + i * 16 + hi * 8 + r;
                        int q   = qt * 32 + j * 16 + l15;
                        if (key > q) st[i][j][r] = -__builtin_inff();
                    }
        }

        // online softmax per q column (lane pair L, L^16 own a column)
#pragma unroll
        for (int j = 0; j < 2; j++) {
            float mx = -__builtin_inff();
#pragma unroll
            for (int i = 0; i < 2; i++)
#pragma unroll
                for (int r = 0; r < 8; r++) mx = fmaxf(mx, st[i][j][r]);
            mx = fmaxf(mx, __shfl_xor(mx, 16, 32));
            float mn = fmaxf(mrow[j], mx);
            float sc = __expf(mrow[j] - mn);
            mrow[j] = mn;
            float sm = 0.0f;
#pragma unroll
            for (int i = 0; i < 2; i++)
#pragma unroll
                for (int r = 0; r < 8; r++) {
                    float e = __expf(st[i][j][r] - mn);
                    st[i][j][r] = e;
                    sm += e;
                }
            sm += __shfl_xor(sm, 16, 32);
            lrow[j] = lrow[j] * sc + sm;
#pragma unroll
            for (int a = 0; a < 4; a++)
#pragma unroll
                for (int r = 0; r < 8; r++) ot[a][j][r] *= sc;
        }

        // rebuild P as B-operand (bf16): cross-half exchange of packed pairs
        FragU pb[2];
#pragma unroll
        for (int j = 0; j < 2; j++) {
            unsigned own[8], oth[8];
#pragma unroll
            for (int i = 0; i < 2; i++)
#pragma unroll
                for (int t = 0; t < 4; t++)
                    own[i * 4 + t] = pack2(st[i][j][2 * t], st[i][j][2 * t + 1]);
#pragma unroll
            for (int u = 0; u < 8; u++) oth[u] = __shfl_xor(own[u], 16, 32);
            unsigned bu[8];
#pragma unroll
            for (int t = 0; t < 4; t++) {
                bu[t]     = hi ? oth[4 + t] : own[t];       // keys 0-7 / 16-23
                bu[4 + t] = hi ? own[4 + t] : oth[t];       // keys 8-15 / 24-31
            }
            pb[j].q[0] = make_uint4(bu[0], bu[1], bu[2], bu[3]);
            pb[j].q[1] = make_uint4(bu[4], bu[5], bu[6], bu[7]);
        }

        // O^T += V^T x P
#pragma unroll
        for (int a = 0; a < 4; a++) {
            FragU av = loadA(vp + (size_t)(a * 16) * S_LEN + kt * 32, S_LEN, lane);
#pragma unroll
            for (int j = 0; j < 2; j++)
                ot[a][j] = wmma_bf16(av.v, pb[j].v, ot[a][j]);
        }
    }

    // epilogue: normalize, pack 8 consecutive d-values per lane -> b128 stores
    int b = bh >> 4, h = bh & (NHEADS - 1);
#pragma unroll
    for (int j = 0; j < 2; j++) {
        float rl = 1.0f / lrow[j];
        int q = qt * 32 + j * 16 + l15;
        unsigned short* op = attn + ((size_t)(b * S_LEN) + q) * D_MODEL + h * DHEAD;
#pragma unroll
        for (int a = 0; a < 4; a++) {
            uint4 w;
            w.x = pack2(ot[a][j][0] * rl, ot[a][j][1] * rl);
            w.y = pack2(ot[a][j][2] * rl, ot[a][j][3] * rl);
            w.z = pack2(ot[a][j][4] * rl, ot[a][j][5] * rl);
            w.w = pack2(ot[a][j][6] * rl, ot[a][j][7] * rl);
            *(uint4*)(op + a * 16 + hi * 8) = w;
        }
    }
}

extern "C" void kernel_launch(void* const* d_in, const int* in_sizes, int n_in,
                              void* d_out, int out_size, void* d_ws, size_t ws_size,
                              hipStream_t stream) {
    (void)in_sizes; (void)n_in; (void)out_size; (void)ws_size;
    const float* x  = (const float*)d_in[0];
    const int*   tp = (const int*)d_in[1];
    const float* Wq = (const float*)d_in[2];
    const float* Wk = (const float*)d_in[3];
    const float* Wv = (const float*)d_in[4];
    const float* Wo = (const float*)d_in[5];

    // workspace layout (bf16 = 2 bytes), total ~88 MB
    char* ws = (char*)d_ws;
    unsigned short* xb  = (unsigned short*)ws;  ws += (size_t)M_ROWS * D_MODEL * 2;   // 16 MB
    unsigned short* wqb = (unsigned short*)ws;  ws += (size_t)D_MODEL * D_MODEL * 2;  // 2 MB
    unsigned short* wkb = (unsigned short*)ws;  ws += (size_t)D_MODEL * D_MODEL * 2;
    unsigned short* wvb = (unsigned short*)ws;  ws += (size_t)D_MODEL * D_MODEL * 2;
    unsigned short* wob = (unsigned short*)ws;  ws += (size_t)D_MODEL * D_MODEL * 2;
    unsigned short* qb  = (unsigned short*)ws;  ws += (size_t)M_ROWS * D_MODEL * 2;   // [B,H,S,Dh]
    unsigned short* kb  = (unsigned short*)ws;  ws += (size_t)M_ROWS * D_MODEL * 2;   // [B,H,S,Dh]
    unsigned short* vtb = (unsigned short*)ws;  ws += (size_t)M_ROWS * D_MODEL * 2;   // [B,H,Dh,S]
    unsigned short* ab  = (unsigned short*)ws;  ws += (size_t)M_ROWS * D_MODEL * 2;   // [B,S,D]

    int n4x = M_ROWS * D_MODEL / 4;
    int n4w = D_MODEL * D_MODEL / 4;
    cvt_f32_bf16<<<n4x / 256, 256, 0, stream>>>(x,  xb,  n4x);
    cvt_f32_bf16<<<n4w / 256, 256, 0, stream>>>(Wq, wqb, n4w);
    cvt_f32_bf16<<<n4w / 256, 256, 0, stream>>>(Wk, wkb, n4w);
    cvt_f32_bf16<<<n4w / 256, 256, 0, stream>>>(Wv, wvb, n4w);
    cvt_f32_bf16<<<n4w / 256, 256, 0, stream>>>(Wo, wob, n4w);

    dim3 gg(D_MODEL / 128, M_ROWS / 128);   // (8, 64)
    gemm_bf16<0><<<gg, 256, 0, stream>>>(xb, wqb, (void*)qb);
    gemm_bf16<0><<<gg, 256, 0, stream>>>(xb, wkb, (void*)kb);
    gemm_bf16<1><<<gg, 256, 0, stream>>>(xb, wvb, (void*)vtb);

    int nr = BATCH * NHEADS * S_LEN * (DHEAD / 2);
    rope_kernel<<<nr / 256, 256, 0, stream>>>(qb, tp, 0.125f);   // fold 1/sqrt(64)
    rope_kernel<<<nr / 256, 256, 0, stream>>>(kb, tp, 1.0f);

    flash_attn<<<dim3(S_LEN / 32, BATCH * NHEADS), 32, 0, stream>>>(qb, kb, vtb, ab);

    gemm_bf16<2><<<gg, 256, 0, stream>>>(ab, wob, d_out);
}
